// ResidualBlockNoBNDynamic_drt_87806311399738
// MI455X (gfx1250) — matrix operand order repaired and checked by reading, not compiled, not run on hardware
//
#include <hip/hip_runtime.h>
#include <stdint.h>
#include <stddef.h>

// Problem constants (from reference)
#define BB 8
#define CC 64
#define HH 256
#define WWID 256
#define NE 10

typedef __bf16 v16bf __attribute__((ext_vector_type(16)));
typedef __bf16 v8bf  __attribute__((ext_vector_type(8)));
typedef float  v8f   __attribute__((ext_vector_type(8)));
typedef float  v4f   __attribute__((ext_vector_type(4)));

static __device__ __forceinline__ v8f zero8f() {
  v8f z;
#pragma unroll
  for (int i = 0; i < 8; ++i) z[i] = 0.0f;
  return z;
}

// Load 16 bf16 (32B) as two 16B chunks -> one v16bf (works for LDS and global,
// only requires 16B alignment).
static __device__ __forceinline__ v16bf ld16bf(const __bf16* p) {
  v8bf lo = *(const v8bf*)p;
  v8bf hi = *(const v8bf*)(p + 8);
  return __builtin_shufflevector(lo, hi, 0,1,2,3,4,5,6,7,8,9,10,11,12,13,14,15);
}

// ---------------------------------------------------------------------------
// Prep: aggregate per-batch 3x3 expert weights into WMMA A-fragment order.
// dst layout: [b][mo(4)][kc(18)][lane(32)][e(16)] bf16
// A K-order used here, matching B (16-bit A 16x32): K = 16*g + e, g=lane>>4
// Logical k = tap*64 + cin,  kc covers k in [kc*32, kc*32+32)
// ---------------------------------------------------------------------------
__global__ void k_prep_conv_frags(const float* __restrict__ attn, // [B,NE]
                                  const float* __restrict__ wk,   // [NE,64,64,3,3]
                                  __bf16* __restrict__ dst) {
  int tid = blockIdx.x * blockDim.x + threadIdx.x;
  if (tid >= BB * 4 * 18 * 32) return;
  int lane = tid & 31;
  int kc   = (tid >> 5) % 18;
  int mo   = (tid / (32 * 18)) & 3;
  int b    = tid / (32 * 18 * 4);
  int g    = lane >> 4;
  int o    = mo * 16 + (lane & 15);
  float aw[NE];
#pragma unroll
  for (int kk = 0; kk < NE; ++kk) aw[kk] = attn[b * NE + kk];
  __bf16* out = dst + (size_t)tid * 16;
#pragma unroll
  for (int e = 0; e < 16; ++e) {
    int k = kc * 32 + 16 * g + e;
    int tp = k >> 6;          // tap 0..8 = dy*3+dx
    int c  = k & 63;          // input channel
    float s = 0.0f;
#pragma unroll
    for (int kk = 0; kk < NE; ++kk)
      s += aw[kk] * wk[(((size_t)kk * 64 + o) * 64 + c) * 9 + tp];
    out[e] = (__bf16)s;
  }
}

// Prep: 1x1 weights (3 of them) into fragment order. dst: [j(3)][mo(4)][kc(2)][32][16]
__global__ void k_prep_1x1_frags(const float* __restrict__ w0,
                                 const float* __restrict__ w1,
                                 const float* __restrict__ w2,
                                 __bf16* __restrict__ dst) {
  int tid = blockIdx.x * blockDim.x + threadIdx.x;
  if (tid >= 3 * 4 * 2 * 32) return;
  int lane = tid & 31;
  int kc   = (tid >> 5) & 1;
  int mo   = (tid >> 6) & 3;
  int j    = tid >> 8;
  const float* wj = (j == 0) ? w0 : ((j == 1) ? w1 : w2);
  int g = lane >> 4;
  int o = mo * 16 + (lane & 15);
  __bf16* out = dst + (size_t)tid * 16;
#pragma unroll
  for (int e = 0; e < 16; ++e) {
    int k = kc * 32 + 16 * g + e; // k == cin (K=64)
    out[e] = (__bf16)wj[o * 64 + k];
  }
}

// Prep: aggregated bias[b][o] = sum_k attn[b,k] * bk[k][o]
__global__ void k_prep_bias(const float* __restrict__ attn,
                            const float* __restrict__ bk,
                            float* __restrict__ dst) {
  int tid = blockIdx.x * blockDim.x + threadIdx.x;
  if (tid >= BB * CC) return;
  int b = tid / CC, o = tid & 63;
  float s = 0.0f;
#pragma unroll
  for (int kk = 0; kk < NE; ++kk) s += attn[b * NE + kk] * bk[kk * CC + o];
  dst[tid] = s;
}

// ---------------------------------------------------------------------------
// Tile geometry for both conv stages: output tile 8 rows x 32 cols, halo 1.
// LDS tile: 10 rows x 34 cols x 64 channels (padded pitch 72) in bf16 NHWC.
// Block: 256 threads = 8 waves; wave = (mo 0..3, half 0..1);
// each wave owns 8 WMMA N-tiles (4 rows x 2 column-tiles of 16 pixels).
// ---------------------------------------------------------------------------
#define TR 10
#define TC 34
#define CP 72

// Stage 1: out1 = relu((conv3x3_A2(x) + bias2)*gamma + sum_j par_j * (W1x1_j x))
__global__ __launch_bounds__(256) void k_stage1(
    const float*  __restrict__ x,     // [B,64,H,W] fp32
    const float*  __restrict__ par,   // [B,3,1,H,W]
    const float*  __restrict__ gamma, // [B,64]
    const __bf16* __restrict__ Afrag, // [B,4,18,512]
    const __bf16* __restrict__ Wfrag, // [3,4,2,512]
    const float*  __restrict__ bias,  // [B,64]
    __bf16*       __restrict__ out1)  // [B,H,W,64] bf16
{
  __shared__ __bf16 tile[TR * TC * CP];
  const int b  = blockIdx.z;
  const int h0 = blockIdx.y * 8;
  const int w0 = blockIdx.x * 32;
  const int tid = threadIdx.x;

  // Load fp32 NCHW -> bf16 NHWC LDS tile (transpose), zero-pad borders.
  for (int i = tid; i < CC * TR * TC; i += 256) {
    int c   = i / (TR * TC);
    int rem = i - c * (TR * TC);
    int r   = rem / TC;
    int col = rem - r * TC;
    int hy = h0 - 1 + r, wx = w0 - 1 + col;
    float v = 0.0f;
    if ((unsigned)hy < (unsigned)HH && (unsigned)wx < (unsigned)WWID)
      v = x[(((size_t)b * CC + c) * HH + hy) * WWID + wx];
    tile[(r * TC + col) * CP + c] = (__bf16)v;
  }
  __syncthreads();

  const int wv = tid >> 5, lane = tid & 31;
  const int mo = wv & 3, half = wv >> 2;
  const int g = lane >> 4, nn = lane & 15;

  v8f acc[8];
#pragma unroll
  for (int q = 0; q < 8; ++q) acc[q] = zero8f();

  const __bf16* Ab = Afrag + (size_t)(b * 4 + mo) * 18 * 512 + lane * 16;
  for (int kc = 0; kc < 18; ++kc) {
    v16bf a = ld16bf(Ab + kc * 512);
    int tp = kc >> 1;
    int ch = (kc & 1) * 32 + g * 16;       // 16 consecutive channels
    int dy = tp / 3 - 1;
    int dx = tp - (tp / 3) * 3 - 1;
#pragma unroll
    for (int nt = 0; nt < 8; ++nt) {
      int rr  = half * 4 + (nt >> 1);
      int x0c = (nt & 1) * 16;
      int row = rr + 1 + dy;
      int colp = x0c + 1 + dx + nn;
      v16bf bb = ld16bf(&tile[(row * TC + colp) * CP + ch]);
      acc[nt] = __builtin_amdgcn_wmma_f32_16x16x32_bf16(
          false, a, false, bb, (short)0, acc[nt], false, false);
    }
  }

  const int m0 = mo * 16 + g * 8;
  v4f bia0 = *(const v4f*)(bias  + b * CC + m0);
  v4f bia1 = *(const v4f*)(bias  + b * CC + m0 + 4);
  v4f gm0  = *(const v4f*)(gamma + b * CC + m0);
  v4f gm1  = *(const v4f*)(gamma + b * CC + m0 + 4);

#pragma unroll
  for (int nt = 0; nt < 8; ++nt) {
    int rr  = half * 4 + (nt >> 1);
    int x0c = (nt & 1) * 16;
    int hh2 = h0 + rr, ww2 = w0 + x0c + nn;

    // dyres: three 1x1 GEMMs on the center pixel, scaled by per-pixel par_j
    v8f dacc = zero8f();
#pragma unroll
    for (int j = 0; j < 3; ++j) {
      v8f t1 = zero8f();
#pragma unroll
      for (int kc = 0; kc < 2; ++kc) {
        v16bf a = ld16bf(Wfrag + (size_t)((j * 4 + mo) * 2 + kc) * 512 + lane * 16);
        int ch = kc * 32 + g * 16;
        v16bf bb = ld16bf(&tile[((rr + 1) * TC + (x0c + 1 + nn)) * CP + ch]);
        t1 = __builtin_amdgcn_wmma_f32_16x16x32_bf16(
            false, a, false, bb, (short)0, t1, false, false);
      }
      float pj = par[(((size_t)b * 3 + j) * HH + hh2) * WWID + ww2];
#pragma unroll
      for (int v = 0; v < 8; ++v) dacc[v] += pj * t1[v];
    }

    v8bf o8;
#pragma unroll
    for (int v = 0; v < 8; ++v) {
      float bv = (v < 4) ? bia0[v] : bia1[v - 4];
      float gv = (v < 4) ? gm0[v]  : gm1[v - 4];
      float r = (acc[nt][v] + bv) * gv + dacc[v];
      r = r > 0.0f ? r : 0.0f;
      o8[v] = (__bf16)r;
    }
    *(v8bf*)(out1 + (((size_t)b * HH + hh2) * WWID + ww2) * CC + m0) = o8;
  }
}

// Stage 2: out = x + (conv3x3_A1(out1) + bias1) * gamma   (fp32 NCHW output)
__global__ __launch_bounds__(256) void k_stage2(
    const __bf16* __restrict__ in1,   // [B,H,W,64] bf16
    const float*  __restrict__ x,     // identity, [B,64,H,W] fp32
    const float*  __restrict__ gamma, // [B,64]
    const __bf16* __restrict__ Afrag, // [B,4,18,512]
    const float*  __restrict__ bias,  // [B,64]
    float*        __restrict__ out)   // [B,64,H,W] fp32
{
  __shared__ __bf16 tile[TR * TC * CP];
  const int b  = blockIdx.z;
  const int h0 = blockIdx.y * 8;
  const int w0 = blockIdx.x * 32;
  const int tid = threadIdx.x;

  // Copy bf16 NHWC tile into LDS (16B chunks; 8 chunks of 8 channels per pixel)
  for (int i = tid; i < TR * TC * 8; i += 256) {
    int cc  = i & 7;
    int col = (i >> 3) % TC;
    int r   = i / (TC * 8);
    int hy = h0 - 1 + r, wx = w0 - 1 + col;
    v8bf v;
#pragma unroll
    for (int e = 0; e < 8; ++e) v[e] = (__bf16)0.0f;
    if ((unsigned)hy < (unsigned)HH && (unsigned)wx < (unsigned)WWID)
      v = *(const v8bf*)(in1 + (((size_t)b * HH + hy) * WWID + wx) * CC + cc * 8);
    *(v8bf*)&tile[(r * TC + col) * CP + cc * 8] = v;
  }
  __syncthreads();

  const int wv = tid >> 5, lane = tid & 31;
  const int mo = wv & 3, half = wv >> 2;
  const int g = lane >> 4, nn = lane & 15;

  v8f acc[8];
#pragma unroll
  for (int q = 0; q < 8; ++q) acc[q] = zero8f();

  const __bf16* Ab = Afrag + (size_t)(b * 4 + mo) * 18 * 512 + lane * 16;
  for (int kc = 0; kc < 18; ++kc) {
    v16bf a = ld16bf(Ab + kc * 512);
    int tp = kc >> 1;
    int ch = (kc & 1) * 32 + g * 16;
    int dy = tp / 3 - 1;
    int dx = tp - (tp / 3) * 3 - 1;
#pragma unroll
    for (int nt = 0; nt < 8; ++nt) {
      int rr  = half * 4 + (nt >> 1);
      int x0c = (nt & 1) * 16;
      int row = rr + 1 + dy;
      int colp = x0c + 1 + dx + nn;
      v16bf bb = ld16bf(&tile[(row * TC + colp) * CP + ch]);
      acc[nt] = __builtin_amdgcn_wmma_f32_16x16x32_bf16(
          false, a, false, bb, (short)0, acc[nt], false, false);
    }
  }

  const int m0 = mo * 16 + g * 8;
  v4f bia0 = *(const v4f*)(bias  + b * CC + m0);
  v4f bia1 = *(const v4f*)(bias  + b * CC + m0 + 4);
  v4f gm0  = *(const v4f*)(gamma + b * CC + m0);
  v4f gm1  = *(const v4f*)(gamma + b * CC + m0 + 4);

#pragma unroll
  for (int nt = 0; nt < 8; ++nt) {
    int rr  = half * 4 + (nt >> 1);
    int x0c = (nt & 1) * 16;
    int hh2 = h0 + rr, ww2 = w0 + x0c + nn;
#pragma unroll
    for (int v = 0; v < 8; ++v) {
      int m = m0 + v;
      size_t xi = (((size_t)b * CC + m) * HH + hh2) * WWID + ww2;
      float bv = (v < 4) ? bia0[v] : bia1[v - 4];
      float gv = (v < 4) ? gm0[v]  : gm1[v - 4];
      out[xi] = x[xi] + (acc[nt][v] + bv) * gv;
    }
  }
}

// ---------------------------------------------------------------------------
// Launch
// ---------------------------------------------------------------------------
extern "C" void kernel_launch(void* const* d_in, const int* in_sizes, int n_in,
                              void* d_out, int out_size, void* d_ws, size_t ws_size,
                              hipStream_t stream) {
  const float* x     = (const float*)d_in[0];   // [8,64,256,256]
  const float* attn  = (const float*)d_in[1];   // [8,10]
  const float* gamma = (const float*)d_in[2];   // [8,64]
  const float* par   = (const float*)d_in[3];   // [8,3,1,256,256]
  const float* w1    = (const float*)d_in[4];   // [10,64,64,3,3]
  const float* b1    = (const float*)d_in[5];   // [10,64]
  const float* w2    = (const float*)d_in[6];   // [10,64,64,3,3]
  const float* b2    = (const float*)d_in[7];   // [10,64]
  const float* w16   = (const float*)d_in[8];   // [64,64]
  const float* w8a   = (const float*)d_in[9];   // [64,64]
  const float* w8b   = (const float*)d_in[10];  // [64,64]

  char* ws = (char*)d_ws;
  const size_t OUT1_B  = (size_t)BB * HH * WWID * CC * 2;  // 67,108,864
  const size_t AFRAG_B = (size_t)BB * 4 * 18 * 512 * 2;    // 589,824
  const size_t W11_B   = (size_t)3 * 4 * 2 * 512 * 2;      // 24,576
  __bf16* out1  = (__bf16*)ws;
  __bf16* A2    = (__bf16*)(ws + OUT1_B);
  __bf16* A1    = (__bf16*)(ws + OUT1_B + AFRAG_B);
  __bf16* W11   = (__bf16*)(ws + OUT1_B + 2 * AFRAG_B);
  float*  bias2 = (float*)(ws + OUT1_B + 2 * AFRAG_B + W11_B);
  float*  bias1 = bias2 + BB * CC;

  k_prep_conv_frags<<<72, 256, 0, stream>>>(attn, w2, A2);
  k_prep_conv_frags<<<72, 256, 0, stream>>>(attn, w1, A1);
  k_prep_1x1_frags<<<3, 256, 0, stream>>>(w16, w8a, w8b, W11);
  k_prep_bias<<<2, 256, 0, stream>>>(attn, b2, bias2);
  k_prep_bias<<<2, 256, 0, stream>>>(attn, b1, bias1);

  dim3 grid(WWID / 32, HH / 8, BB);  // (8, 32, 8)
  k_stage1<<<grid, 256, 0, stream>>>(x, par, gamma, A2, W11, bias2, out1);
  k_stage2<<<grid, 256, 0, stream>>>(out1, x, gamma, A1, bias1, (float*)d_out);
}
